// EdgeAttentionFlat_51728586113551
// MI455X (gfx1250) — matrix operand order repaired and checked by reading, not compiled, not run on hardware
//
#include <hip/hip_runtime.h>
#include <math.h>

#define BB  2
#define NN  128
#define DD  256
#define HH  8
#define DKK 32

typedef __attribute__((ext_vector_type(16))) __bf16 v16bf;
typedef __attribute__((ext_vector_type(8)))  __bf16 v8bf;
typedef __attribute__((ext_vector_type(8)))  float  v8f;

__device__ __forceinline__ __bf16 f2bf(float f) {
  unsigned u = __builtin_bit_cast(unsigned, f);
  unsigned r = u + 0x7FFFu + ((u >> 16) & 1u);   // round-to-nearest-even
  unsigned short h = (unsigned short)(r >> 16);
  return __builtin_bit_cast(__bf16, h);
}
__device__ __forceinline__ float bf2f(__bf16 b) {
  unsigned u = ((unsigned)__builtin_bit_cast(unsigned short, b)) << 16;
  return __builtin_bit_cast(float, u);
}

__device__ __forceinline__ v16bf load_afrag_bf16(const __bf16* p) {
  // A 16x32 bf16 layout: lane(row)=L&15; lo 8 elems at K=8*half, hi 8 at K=16+8*half
  v8bf lo = *(const v8bf*)p;
  v8bf hi = *(const v8bf*)(p + 16);
  v16bf a;
#pragma unroll
  for (int i = 0; i < 8; ++i) { a[i] = lo[i]; a[8 + i] = hi[i]; }
  return a;
}
__device__ __forceinline__ v16bf load_bfrag_bf16(const __bf16* p) {
  // B 32x16 bf16 layout: lane(col)=L&15; lo 8 elems at K=8*half, hi 8 at K=16+8*half
  v8bf b0 = *(const v8bf*)p;
  v8bf b1 = *(const v8bf*)(p + 16);
  v16bf b;
#pragma unroll
  for (int i = 0; i < 8; ++i) { b[i] = b0[i]; b[8 + i] = b1[i]; }
  return b;
}

// ---------------------------------------------------------------- weights cvt
__global__ __launch_bounds__(256) void cvt_f32_bf16(const float* __restrict__ s,
                                                    __bf16* __restrict__ d, int n) {
  int i = blockIdx.x * blockDim.x + threadIdx.x;
  if (i < n) d[i] = f2bf(s[i]);
}

// -------------------------------------------------- Y[M,256] = X[M,256]@W^T
// one 16x16 output tile per wave; K loop 256 in 8 WMMA steps
template <bool A_BF16, bool OUT_F32>
__global__ __launch_bounds__(256) void gemm_nt(const void* __restrict__ Ap,
                                               const __bf16* __restrict__ W,
                                               void* __restrict__ Yp, float scale) {
  const int ln   = threadIdx.x & 31;
  const int wave = threadIdx.x >> 5;
  const int col  = ln & 15;
  const int hf   = ln >> 4;
  const int tile = blockIdx.x * 8 + wave;
  const int tm = tile >> 4;   // 2048 M tiles
  const int tn = tile & 15;   // 16 N tiles
  const long arow = (long)(tm * 16 + col);

  v8f c = {};
#pragma unroll
  for (int k0 = 0; k0 < DD; k0 += 32) {
    v16bf a;
    if (A_BF16) {
      a = load_afrag_bf16((const __bf16*)Ap + arow * DD + k0 + 8 * hf);
    } else {
      const float* ap = (const float*)Ap + arow * DD + k0 + 8 * hf;
      float4 f0 = *(const float4*)(ap);
      float4 f1 = *(const float4*)(ap + 4);
      float4 g0 = *(const float4*)(ap + 16);
      float4 g1 = *(const float4*)(ap + 20);
      a[0] = f2bf(f0.x); a[1] = f2bf(f0.y); a[2] = f2bf(f0.z); a[3] = f2bf(f0.w);
      a[4] = f2bf(f1.x); a[5] = f2bf(f1.y); a[6] = f2bf(f1.z); a[7] = f2bf(f1.w);
      a[8]  = f2bf(g0.x); a[9]  = f2bf(g0.y); a[10] = f2bf(g0.z); a[11] = f2bf(g0.w);
      a[12] = f2bf(g1.x); a[13] = f2bf(g1.y); a[14] = f2bf(g1.z); a[15] = f2bf(g1.w);
    }
    v16bf b = load_bfrag_bf16(W + (long)(tn * 16 + col) * DD + k0 + 8 * hf);
    c = __builtin_amdgcn_wmma_f32_16x16x32_bf16(false, a, false, b, (short)0, c, false, false);
  }
#pragma unroll
  for (int r = 0; r < 8; ++r) {
    long idx = (long)(tm * 16 + r + 8 * hf) * DD + tn * 16 + col;
    float v = c[r] * scale;
    if (OUT_F32) ((float*)Yp)[idx] = v;
    else         ((__bf16*)Yp)[idx] = f2bf(v);
  }
}

// --------------------------------------------- pass1: per-row softmax stats
// mode 0: batch=(b,x,h), rows=y, row stride D     (scores_r)
// mode 1: batch=(b,y,h), rows=x, row stride N*D   (scores_l)
__global__ __launch_bounds__(256) void attn_pass1(const __bf16* __restrict__ Qb,
                                                  const __bf16* __restrict__ Kb,
                                                  const unsigned char* __restrict__ mask,
                                                  float2* __restrict__ statsOut, int mode) {
  const int ln   = threadIdx.x & 31;
  const int wave = threadIdx.x >> 5;
  const int col  = ln & 15;
  const int hf   = ln >> 4;
  const int bi = blockIdx.x;
  const int h  = bi % HH;
  const int bt = bi / HH;
  const int xy = bt % NN;
  const int b  = bt / NN;
  const long rowStride = mode ? (long)NN * DD : (long)DD;
  const long base = mode ? ((long)b * NN * NN + xy) * DD + h * DKK
                         : ((long)(b * NN + xy) * NN) * DD + h * DKK;
  const long maskRow  = mode ? (long)NN * NN : (long)NN;
  const long maskBase = mode ? ((long)b * NN * NN * NN + (long)xy * NN)
                             : ((long)(b * NN + xy) * NN) * NN;
  const int tile0 = wave * 16;

  v16bf a = load_afrag_bf16(Qb + base + (long)(tile0 + col) * rowStride + 8 * hf);

  float s[8][8];
#pragma unroll
  for (int zt = 0; zt < 8; ++zt) {
    v16bf bf = load_bfrag_bf16(Kb + base + (long)(zt * 16 + col) * rowStride + 8 * hf);
    v8f c = {};
    c = __builtin_amdgcn_wmma_f32_16x16x32_bf16(false, a, false, bf, (short)0, c, false, false);
    int z = zt * 16 + col;
#pragma unroll
    for (int r = 0; r < 8; ++r) {
      float v = c[r];
      int row = tile0 + r + 8 * hf;
      if (mask[maskBase + (long)row * maskRow + z]) v = -__builtin_inff();
      s[zt][r] = v;
    }
  }
  float m[8], l[8];
#pragma unroll
  for (int r = 0; r < 8; ++r) {
    float mv = s[0][r];
#pragma unroll
    for (int zt = 1; zt < 8; ++zt) mv = fmaxf(mv, s[zt][r]);
    m[r] = mv;
  }
#pragma unroll
  for (int d = 1; d < 16; d <<= 1)
#pragma unroll
    for (int r = 0; r < 8; ++r) m[r] = fmaxf(m[r], __shfl_xor(m[r], d, 32));
#pragma unroll
  for (int r = 0; r < 8; ++r) {
    float lv = 0.f;
#pragma unroll
    for (int zt = 0; zt < 8; ++zt) lv += __expf(s[zt][r] - m[r]);
    l[r] = lv;
  }
#pragma unroll
  for (int d = 1; d < 16; d <<= 1)
#pragma unroll
    for (int r = 0; r < 8; ++r) l[r] += __shfl_xor(l[r], d, 32);

  if (col == 0) {
#pragma unroll
    for (int r = 0; r < 8; ++r) {
      int row = tile0 + r + 8 * hf;
      float2 st; st.x = m[r]; st.y = l[r];
      statsOut[(long)bi * NN + row] = st;
    }
  }
}

// ------------------------------- pass2: normalized probs x V, accumulate X
__global__ __launch_bounds__(256) void attn_pass2(const __bf16* __restrict__ Qb,
                                                  const __bf16* __restrict__ Kb,
                                                  const __bf16* __restrict__ Vb,
                                                  const float2* __restrict__ statsR,
                                                  const float2* __restrict__ statsL,
                                                  const unsigned char* __restrict__ mask,
                                                  __bf16* __restrict__ X, int mode) {
  __shared__ __attribute__((aligned(16))) __bf16 lds[8][16][40];
  const int ln   = threadIdx.x & 31;
  const int wave = threadIdx.x >> 5;
  const int col  = ln & 15;
  const int hf   = ln >> 4;
  const int bi = blockIdx.x;
  const int h  = bi % HH;
  const int bt = bi / HH;
  const int xy = bt % NN;
  const int b  = bt / NN;
  const long rowStride = mode ? (long)NN * DD : (long)DD;
  const long base = mode ? ((long)b * NN * NN + xy) * DD + h * DKK
                         : ((long)(b * NN + xy) * NN) * DD + h * DKK;
  const long maskRow  = mode ? (long)NN * NN : (long)NN;
  const long maskBase = mode ? ((long)b * NN * NN * NN + (long)xy * NN)
                             : ((long)(b * NN + xy) * NN) * NN;
  const int tile0 = wave * 16;

  // combined softmax stats for this batch's rows (joint over r+l halves)
  const float2* own = (mode ? statsL : statsR) + (long)bi * NN;
  float rowM[8], rowInv[8];
#pragma unroll
  for (int r = 0; r < 8; ++r) {
    int row = tile0 + r + 8 * hf;
    float2 so = own[row];
    long oi = ((long)(b * NN + row) * HH + h) * NN + xy;
    float2 ot = mode ? statsR[oi] : statsL[oi];
    float M = fmaxf(so.x, ot.x);
    float L = so.y * __expf(so.x - M) + ot.y * __expf(ot.x - M);
    rowM[r] = M;
    rowInv[r] = 1.0f / L;
  }

  v16bf a = load_afrag_bf16(Qb + base + (long)(tile0 + col) * rowStride + 8 * hf);

  v8f acc0 = {}, acc1 = {};
  for (int zc = 0; zc < 4; ++zc) {
#pragma unroll
    for (int tt = 0; tt < 2; ++tt) {
      int zt = zc * 2 + tt;
      v16bf bf = load_bfrag_bf16(Kb + base + (long)(zt * 16 + col) * rowStride + 8 * hf);
      v8f c = {};
      c = __builtin_amdgcn_wmma_f32_16x16x32_bf16(false, a, false, bf, (short)0, c, false, false);
      int z = zt * 16 + col;
#pragma unroll
      for (int r = 0; r < 8; ++r) {
        float v = c[r];
        int row = tile0 + r + 8 * hf;
        if (mask[maskBase + (long)row * maskRow + z]) v = -__builtin_inff();
        float p = __expf(v - rowM[r]) * rowInv[r];
        lds[wave][r + 8 * hf][tt * 16 + col] = f2bf(p);
      }
    }
    __syncthreads();
    // prob tile (16 rows x 32 z) -> A fragment
    v16bf ap = load_afrag_bf16(&lds[wave][col][8 * hf]);
#pragma unroll
    for (int tt = 0; tt < 2; ++tt) {
      v16bf vb;
#pragma unroll
      for (int e = 0; e < 16; ++e) {
        int z = zc * 32 + ((e < 8) ? 8 * hf + e : 16 + 8 * hf + (e - 8));
        vb[e] = Vb[base + (long)z * rowStride + tt * 16 + col];
      }
      if (tt == 0)
        acc0 = __builtin_amdgcn_wmma_f32_16x16x32_bf16(false, ap, false, vb, (short)0, acc0, false, false);
      else
        acc1 = __builtin_amdgcn_wmma_f32_16x16x32_bf16(false, ap, false, vb, (short)0, acc1, false, false);
    }
    __syncthreads();
  }
#pragma unroll
  for (int tt = 0; tt < 2; ++tt) {
#pragma unroll
    for (int r = 0; r < 8; ++r) {
      long idx = base + (long)(tile0 + r + 8 * hf) * rowStride + tt * 16 + col;
      float v = (tt == 0) ? acc0[r] : acc1[r];
      if (mode) v += bf2f(X[idx]);   // pass2-l accumulates onto pass2-r result
      X[idx] = f2bf(v);
    }
  }
}

extern "C" void kernel_launch(void* const* d_in, const int* in_sizes, int n_in,
                              void* d_out, int out_size, void* d_ws, size_t ws_size,
                              hipStream_t stream) {
  (void)in_sizes; (void)n_in; (void)out_size; (void)ws_size;
  const float* query = (const float*)d_in[0];
  const float* key   = (const float*)d_in[1];
  const float* value = (const float*)d_in[2];
  const unsigned char* mask = (const unsigned char*)d_in[3];
  const float* Wk = (const float*)d_in[4];
  const float* Wv = (const float*)d_in[5];
  const float* Wq = (const float*)d_in[6];
  const float* Wo = (const float*)d_in[7];

  char* ws = (char*)d_ws;
  size_t off = 0;
  auto alloc = [&](size_t bytes) {
    char* p = ws + off;
    off += (bytes + 255) & ~(size_t)255;
    return p;
  };
  __bf16* Wqb = (__bf16*)alloc((size_t)DD * DD * 2);
  __bf16* Wkb = (__bf16*)alloc((size_t)DD * DD * 2);
  __bf16* Wvb = (__bf16*)alloc((size_t)DD * DD * 2);
  __bf16* Wob = (__bf16*)alloc((size_t)DD * DD * 2);
  const size_t TSZ = (size_t)BB * NN * NN * DD;       // 8,388,608 elems
  __bf16* Qb = (__bf16*)alloc(TSZ * 2);
  __bf16* Kb = (__bf16*)alloc(TSZ * 2);
  __bf16* Vb = (__bf16*)alloc(TSZ * 2);
  __bf16* Xb = (__bf16*)alloc(TSZ * 2);
  const size_t NSTAT = (size_t)BB * NN * HH * NN;     // 262,144 rows
  float2* statsR = (float2*)alloc(NSTAT * sizeof(float2));
  float2* statsL = (float2*)alloc(NSTAT * sizeof(float2));

  const float inv = 0.17677669529663689f;  // 1/sqrt(DK)

  cvt_f32_bf16<<<256, 256, 0, stream>>>(Wq, Wqb, DD * DD);
  cvt_f32_bf16<<<256, 256, 0, stream>>>(Wk, Wkb, DD * DD);
  cvt_f32_bf16<<<256, 256, 0, stream>>>(Wv, Wvb, DD * DD);
  cvt_f32_bf16<<<256, 256, 0, stream>>>(Wo, Wob, DD * DD);

  // projections: 32768x256 @ 256x256^T, 1/sqrt(DK) folded into Q
  gemm_nt<false, false><<<4096, 256, 0, stream>>>(query, Wqb, Qb, inv);
  gemm_nt<false, false><<<4096, 256, 0, stream>>>(key,   Wkb, Kb, 1.0f);
  gemm_nt<false, false><<<4096, 256, 0, stream>>>(value, Wvb, Vb, 1.0f);

  // softmax row stats: r-batches then l-batches (B*N*H = 2048 each)
  attn_pass1<<<2048, 256, 0, stream>>>(Qb, Kb, mask, statsR, 0);
  attn_pass1<<<2048, 256, 0, stream>>>(Qb, Kb, mask, statsL, 1);

  // normalized attention x V: r writes X, l accumulates (stream-ordered)
  attn_pass2<<<2048, 256, 0, stream>>>(Qb, Kb, Vb, statsR, statsL, mask, Xb, 0);
  attn_pass2<<<2048, 256, 0, stream>>>(Qb, Kb, Vb, statsR, statsL, mask, Xb, 1);

  // output projection to fp32 d_out
  gemm_nt<true, true><<<4096, 256, 0, stream>>>(Xb, Wob, d_out, 1.0f);
}
